// GNN_13365938225711
// MI455X (gfx1250) — matrix operand order repaired and checked by reading, not compiled, not run on hardware
//
#include <hip/hip_runtime.h>
#include <hip/hip_bf16.h>
#include <math.h>

// ---------------- problem constants (match reference) ----------------
#define NN      262144      // total nodes
#define GG      1024        // graphs
#define NPG     256         // nodes per graph
#define KNB     10          // knn neighbors
#define EE      (NN*KNB)    // edges
#define HID     20          // MLP hidden width
#define EPS     1e-5f

typedef _Float16 f16;
typedef __attribute__((ext_vector_type(16))) _Float16 v16h;
typedef __attribute__((ext_vector_type(8)))  _Float16 h8;
typedef __attribute__((ext_vector_type(8)))  float    v8f;

// ---------------- WMMA fragment helpers (wave32 layouts per CDNA5 ISA) ----
// A (16x32 f16): lanes 0-15 rows M=0..15, K {0..7,16..23}; lanes 16-31 same rows, K {8..15,24..31}.
// Each lane's 16 slots are two contiguous 16-byte runs -> two b128 loads.
__device__ __forceinline__ v16h loadA(const f16* __restrict__ base, int ld, int lane) {
    int half = lane >> 4, r = lane & 15;
    const f16* rp = base + r * ld + half * 8;
    h8 lo = *(const h8*)(rp);        // K = half*8 .. +7      (slots 0..7)
    h8 hi = *(const h8*)(rp + 16);   // K = 16+half*8 .. +7   (slots 8..15)
    v16h a;
#pragma unroll
    for (int i = 0; i < 8; i++) { a[i] = lo[i]; a[i + 8] = hi[i]; }
    return a;
}

// B fragments are pre-packed lane-major by prep_weights: frag[lane][16] f16 -> two b128 loads.
__device__ __forceinline__ v16h loadBpk(const f16* __restrict__ frag, int lane) {
    const h8* p = (const h8*)(frag + lane * 16);
    h8 lo = p[0], hi = p[1];
    v16h b;
#pragma unroll
    for (int i = 0; i < 8; i++) { b[i] = lo[i]; b[i + 8] = hi[i]; }
    return b;
}

__device__ __forceinline__ v8f wmma_f16(v16h a, v16h b, v8f c) {
    return __builtin_amdgcn_wmma_f32_16x16x32_f16(false, a, false, b, (short)0, c, false, false);
}

__device__ __forceinline__ float lrelu(float x) { return x > 0.f ? x : 0.1f * x; }

// ---------------- tiny utility kernels ----------------
__global__ void zero_buf(float* p, int n) {
    int i = blockIdx.x * blockDim.x + threadIdx.x;
    if (i < n) p[i] = 0.f;
}

// ---------------- weight packing: per conv, 7 lane-major B fragments ----------------
// layout per conv (f16 offsets): Wa0:0 Wa1:512 Wb0:1024 Wb1:1536 W20:2048 W21:2560 W30:3072 (total 3584)
__device__ void pack_frag16(const float* M, int ld, int colbase, f16* dst) {
    for (int t = threadIdx.x; t < 512; t += blockDim.x) {
        int lane = t >> 4, i = t & 15;
        int k = (i < 8) ? 8 * (lane >> 4) + i : 16 + 8 * (lane >> 4) + (i - 8);
        int col = colbase + (lane & 15);
        dst[t] = (f16)M[k * ld + col];
    }
}

__device__ void conv_pack(const float* w1, int inF, const float* w2,
                          const float* w3, int outF, float* M, f16* dst) {
    // Wa = W1a - W1b (applied to h_i)
    for (int t = threadIdx.x; t < 1024; t += blockDim.x) {
        int k = t >> 5, c = t & 31;
        M[t] = (k < inF && c < HID) ? (w1[k * HID + c] - w1[(inF + k) * HID + c]) : 0.f;
    }
    __syncthreads();
    pack_frag16(M, 32, 0, dst);  pack_frag16(M, 32, 16, dst + 512);
    __syncthreads();
    // Wb = W1b (applied to h_j)
    for (int t = threadIdx.x; t < 1024; t += blockDim.x) {
        int k = t >> 5, c = t & 31;
        M[t] = (k < inF && c < HID) ? w1[(inF + k) * HID + c] : 0.f;
    }
    __syncthreads();
    pack_frag16(M, 32, 0, dst + 1024);  pack_frag16(M, 32, 16, dst + 1536);
    __syncthreads();
    // W2 (20x20 padded)
    for (int t = threadIdx.x; t < 1024; t += blockDim.x) {
        int k = t >> 5, c = t & 31;
        M[t] = (k < HID && c < HID) ? w2[k * HID + c] : 0.f;
    }
    __syncthreads();
    pack_frag16(M, 32, 0, dst + 2048);  pack_frag16(M, 32, 16, dst + 2560);
    __syncthreads();
    // W3 (20xoutF padded to 32x16)
    for (int t = threadIdx.x; t < 512; t += blockDim.x) {
        int k = t >> 4, c = t & 15;
        M[t] = (k < HID && c < outF) ? w3[k * outF + c] : 0.f;
    }
    __syncthreads();
    pack_frag16(M, 16, 0, dst + 3072);
    __syncthreads();
}

__global__ __launch_bounds__(256) void prep_weights(
        const float* w1a, const float* w2a, const float* w3a,
        const float* w1b, const float* w2b, const float* w3b,
        const float* w1c, const float* w2c, const float* w3c, f16* wbuf) {
    __shared__ float M[1024];
    conv_pack(w1a, 4,  w2a, w3a, 10, M, wbuf);
    conv_pack(w1b, 10, w2b, w3b, 10, M, wbuf + 3584);
    conv_pack(w1c, 10, w2c, w3c, 5,  M, wbuf + 7168);
}

// h0 = [x, pos] padded to 32 f16 per node
__global__ void build_h0(const float* __restrict__ x, const float* __restrict__ pos,
                         f16* __restrict__ h0p) {
    int idx = blockIdx.x * blockDim.x + threadIdx.x;
    if (idx >= NN * 32) return;
    int n = idx >> 5, c = idx & 31;
    float v = 0.f;
    if (c == 0)     v = x[n];
    else if (c < 4) v = pos[n * 3 + (c - 1)];
    h0p[idx] = (f16)v;
}

// ---------------- KNN: one workgroup per graph ----------------
__global__ __launch_bounds__(NPG) void knn_kernel(const float* __restrict__ feat,
                                                  int F, int ld, int* __restrict__ srcOut) {
    __shared__ float sf[NPG * 10];   // channel-major: sf[c*256 + node]
    int g = blockIdx.x, tid = threadIdx.x;
    const float* fg = feat + (size_t)g * NPG * ld;
    float my[10];
    for (int c = 0; c < F; c++) { my[c] = fg[tid * ld + c]; sf[c * NPG + tid] = my[c]; }
    __syncthreads();
    float bd[KNB]; int bi[KNB];
#pragma unroll
    for (int k = 0; k < KNB; k++) { bd[k] = INFINITY; bi[k] = tid; }
    for (int j = 0; j < NPG; j++) {
        if (j == tid) continue;
        float d = 0.f;
        for (int c = 0; c < F; c++) { float t = my[c] - sf[c * NPG + j]; d += t * t; }
        if (d < bd[KNB - 1]) {
            int p = KNB - 1;
            while (p > 0 && bd[p - 1] > d) { bd[p] = bd[p - 1]; bi[p] = bi[p - 1]; p--; }
            bd[p] = d; bi[p] = j;
        }
    }
    int nglob = g * NPG + tid;
#pragma unroll
    for (int k = 0; k < KNB; k++) srcOut[nglob * KNB + k] = g * NPG + bi[k];
}

// ---------------- node-level GEMMs: U = H*(W1a-W1b), V = H*W1b ----------------
// D fragments staged through LDS, then copied out as contiguous b128 stores
// (a 16-node tile is a contiguous 1KB block of the row-major U/V arrays).
__global__ __launch_bounds__(256) void node_gemm(const f16* __restrict__ hp,
                                                 const f16* __restrict__ WF,
                                                 f16* __restrict__ u, f16* __restrict__ v) {
    __shared__ alignas(16) f16 st[8][16 * 32];
    int wv = threadIdx.x >> 5, lane = threadIdx.x & 31;
    int node0 = (blockIdx.x * 8 + wv) * 16;
    v16h a = loadA(hp + (size_t)node0 * 32, 32, lane);
    v8f z = {};
    v8f u0 = wmma_f16(a, loadBpk(WF + 0,    lane), z);
    v8f u1 = wmma_f16(a, loadBpk(WF + 512,  lane), z);
    v8f v0 = wmma_f16(a, loadBpk(WF + 1024, lane), z);
    v8f v1 = wmma_f16(a, loadBpk(WF + 1536, lane), z);
    f16* T = st[wv];
    int col = lane & 15, rowadd = (lane >> 4) * 8;
#pragma unroll
    for (int r = 0; r < 8; r++) {
        int row = r + rowadd;
        T[row * 32 + col]      = (f16)u0[r];
        T[row * 32 + col + 16] = (f16)u1[r];
    }
    __syncthreads();
    {
        h8* dst = (h8*)(u + (size_t)node0 * 32);
        const h8* s = (const h8*)T;
        dst[lane * 2] = s[lane * 2]; dst[lane * 2 + 1] = s[lane * 2 + 1];
    }
    __syncthreads();
#pragma unroll
    for (int r = 0; r < 8; r++) {
        int row = r + rowadd;
        T[row * 32 + col]      = (f16)v0[r];
        T[row * 32 + col + 16] = (f16)v1[r];
    }
    __syncthreads();
    {
        h8* dst = (h8*)(v + (size_t)node0 * 32);
        const h8* s = (const h8*)T;
        dst[lane * 2] = s[lane * 2]; dst[lane * 2 + 1] = s[lane * 2 + 1];
    }
}

// ---------------- layer-1 BN stats: m1 = U[dst]+V[src]+b1 ----------------
__global__ __launch_bounds__(256) void edge_stats1(const f16* __restrict__ u,
                                                   const f16* __restrict__ v,
                                                   const int* __restrict__ src,
                                                   const float* __restrict__ b1,
                                                   float* __restrict__ stats) {
    __shared__ float ssum[32], ssq[32];
    int tid = threadIdx.x;
    if (tid < 32) { ssum[tid] = 0.f; ssq[tid] = 0.f; }
    __syncthreads();
    int n = blockIdx.x * blockDim.x + tid;
    const h8* up = (const h8*)(u + (size_t)n * 32);
    h8 ua = up[0], ub = up[1], uc = up[2];            // cols 0..23 (20..23 are zero pads)
    float uu[HID], ps[HID], pq[HID];
#pragma unroll
    for (int c = 0; c < HID; c++) {
        float uv = (c < 8) ? (float)ua[c] : (c < 16) ? (float)ub[c - 8] : (float)uc[c - 16];
        uu[c] = uv + b1[c]; ps[c] = 0.f; pq[c] = 0.f;
    }
#pragma unroll
    for (int k = 0; k < KNB; k++) {
        int s = src[n * KNB + k];
        const h8* vp = (const h8*)(v + (size_t)s * 32);
        h8 va = vp[0], vb = vp[1], vc = vp[2];
#pragma unroll
        for (int c = 0; c < HID; c++) {
            float vv = (c < 8) ? (float)va[c] : (c < 16) ? (float)vb[c - 8] : (float)vc[c - 16];
            float m = uu[c] + vv;
            ps[c] += m; pq[c] += m * m;
        }
    }
#pragma unroll
    for (int c = 0; c < HID; c++) { atomicAdd(&ssum[c], ps[c]); atomicAdd(&ssq[c], pq[c]); }
    __syncthreads();
    if (tid < HID) { atomicAdd(&stats[tid], ssum[tid]); atomicAdd(&stats[32 + tid], ssq[tid]); }
}

__global__ void bn_finalize(const float* __restrict__ stats, const float* __restrict__ g,
                            const float* __restrict__ be, float cnt, float* __restrict__ coef) {
    int c = threadIdx.x;
    if (c >= HID) return;
    float mean = stats[c] / cnt;
    float var  = stats[32 + c] / cnt - mean * mean;
    float a = g[c] * rsqrtf(var + EPS);
    coef[c] = a; coef[32 + c] = be[c] - mean * a;
}

// build activated layer-1 tile in LDS: y1[80][32] f16 (zeros in pad cols).
// 160 threads = one (edge, 16-col half) each; all loads/stores are b128.
__device__ void build_y1(f16* __restrict__ y1, int node0,
                         const f16* __restrict__ u, const f16* __restrict__ v,
                         const int* __restrict__ src,
                         const float* __restrict__ b1, const float* __restrict__ coef1) {
    int t = threadIdx.x;            // 0..159
    int e = t >> 1, hh = t & 1, cb = hh * 16;
    int dst = node0 + e / KNB;
    int s   = src[node0 * KNB + e];
    const h8* up = (const h8*)(u + (size_t)dst * 32 + cb);
    const h8* vp = (const h8*)(v + (size_t)s   * 32 + cb);
    h8 u0 = up[0], u1 = up[1], v0 = vp[0], v1 = vp[1];
    h8 o0, o1;
#pragma unroll
    for (int i = 0; i < 8; i++) {
        int c0 = cb + i, c1 = cb + 8 + i;
        float r0 = 0.f, r1 = 0.f;
        if (c0 < HID) {
            float m = (float)u0[i] + (float)v0[i] + b1[c0];
            r0 = lrelu(coef1[c0] * m + coef1[32 + c0]);
        }
        if (c1 < HID) {
            float m = (float)u1[i] + (float)v1[i] + b1[c1];
            r1 = lrelu(coef1[c1] * m + coef1[32 + c1]);
        }
        o0[i] = (f16)r0; o1[i] = (f16)r1;
    }
    h8* op = (h8*)(y1 + e * 32 + cb);
    op[0] = o0; op[1] = o1;
}

// ---------------- layer-2 BN stats: m2 = y1*W2 + b2 via WMMA ----------------
__global__ __launch_bounds__(160) void edge_l2_stats(const f16* __restrict__ u,
                                                     const f16* __restrict__ v,
                                                     const int* __restrict__ src,
                                                     const float* __restrict__ b1,
                                                     const float* __restrict__ coef1,
                                                     const f16* __restrict__ WF,
                                                     const float* __restrict__ b2,
                                                     float* __restrict__ stats) {
    __shared__ alignas(16) f16 y1[80 * 32];
    __shared__ float ssum[32], ssq[32];
    int tid = threadIdx.x;
    if (tid < 32) { ssum[tid] = 0.f; ssq[tid] = 0.f; }
    int node0 = blockIdx.x * 8;
    build_y1(y1, node0, u, v, src, b1, coef1);
    __syncthreads();
    int wv = tid >> 5, lane = tid & 31;
    v16h a = loadA(y1 + wv * 16 * 32, 32, lane);
    v8f z = {};
    v8f m0 = wmma_f16(a, loadBpk(WF + 2048, lane), z);
    v8f m1 = wmma_f16(a, loadBpk(WF + 2560, lane), z);
    int col = lane & 15;
    float s0 = 0.f, q0 = 0.f, s1 = 0.f, q1 = 0.f;
#pragma unroll
    for (int r = 0; r < 8; r++) {
        float a0 = m0[r] + b2[col];
        s0 += a0; q0 += a0 * a0;
        if (col + 16 < HID) {
            float a1 = m1[r] + b2[col + 16];
            s1 += a1; q1 += a1 * a1;
        }
    }
    atomicAdd(&ssum[col], s0); atomicAdd(&ssq[col], q0);
    if (col + 16 < HID) { atomicAdd(&ssum[col + 16], s1); atomicAdd(&ssq[col + 16], q1); }
    __syncthreads();
    if (tid < HID) { atomicAdd(&stats[tid], ssum[tid]); atomicAdd(&stats[32 + tid], ssq[tid]); }
}

// ---------------- full edge MLP + fused segment-max (8 nodes / block) ----------------
__global__ __launch_bounds__(160) void edge_l3_aggr(const f16* __restrict__ u,
                                                    const f16* __restrict__ v,
                                                    const int* __restrict__ src,
                                                    const float* __restrict__ b1,
                                                    const float* __restrict__ coef1,
                                                    const f16* __restrict__ WF,
                                                    const float* __restrict__ b2,
                                                    const float* __restrict__ coef2,
                                                    const float* __restrict__ b3,
                                                    int outF,
                                                    f16* __restrict__ out16,
                                                    float* __restrict__ out32) {
    __shared__ alignas(16) f16 y1[80 * 32];
    __shared__ float m3s[80 * 16];
    int tid = threadIdx.x;
    int node0 = blockIdx.x * 8;
    build_y1(y1, node0, u, v, src, b1, coef1);
    __syncthreads();
    int wv = tid >> 5, lane = tid & 31;
    f16* myt = y1 + wv * 16 * 32;
    v16h a = loadA(myt, 32, lane);
    v8f z = {};
    v8f m0 = wmma_f16(a, loadBpk(WF + 2048, lane), z);
    v8f m1 = wmma_f16(a, loadBpk(WF + 2560, lane), z);
    int col = lane & 15, rowadd = (lane >> 4) * 8;
#pragma unroll
    for (int r = 0; r < 8; r++) {
        int row = r + rowadd;
        float t0 = lrelu(coef2[col] * (m0[r] + b2[col]) + coef2[32 + col]);
        float t1 = 0.f;
        if (col + 16 < HID)
            t1 = lrelu(coef2[col + 16] * (m1[r] + b2[col + 16]) + coef2[32 + col + 16]);
        myt[row * 32 + col]      = (f16)t0;
        myt[row * 32 + col + 16] = (f16)t1;
    }
    __syncthreads();
    v16h a2 = loadA(myt, 32, lane);
    v8f m3 = wmma_f16(a2, loadBpk(WF + 3072, lane), z);
#pragma unroll
    for (int r = 0; r < 8; r++) {
        int row = r + rowadd;
        m3s[(wv * 16 + row) * 16 + col] = m3[r] + (col < outF ? b3[col] : 0.f);
    }
    __syncthreads();
    if (tid < 128) {
        int j = tid >> 4, c = tid & 15;
        float mx = -INFINITY;
#pragma unroll
        for (int i = 0; i < KNB; i++) mx = fmaxf(mx, m3s[(j * KNB + i) * 16 + c]);
        int node = node0 + j;
        float w = (c < outF) ? mx : 0.f;
        if (out32) out32[node * 16 + c] = w;
        if (out16) { out16[node * 32 + c] = (f16)w; out16[node * 32 + c + 16] = (f16)0.f; }
    }
}

// ---------------- per-graph sum pool ----------------
__global__ __launch_bounds__(NPG) void pool_kernel(const float* __restrict__ h3,
                                                   float* __restrict__ pooled) {
    __shared__ float acc[16];
    int g = blockIdx.x, tid = threadIdx.x;
    if (tid < 16) acc[tid] = 0.f;
    __syncthreads();
    int node = g * NPG + tid;
#pragma unroll
    for (int c = 0; c < 5; c++) atomicAdd(&acc[c], h3[node * 16 + c]);
    __syncthreads();
    if (tid < 16) pooled[g * 16 + tid] = acc[tid];
}

// ---------------- final FFN (BN over 1024 graphs) in one block ----------------
__device__ void bn_act_block(float* h, const float* g, const float* be,
                             float* ssum, float* ssq, float* coef) {
    int tid = threadIdx.x;
    if (tid < HID) { ssum[tid] = 0.f; ssq[tid] = 0.f; }
    __syncthreads();
#pragma unroll
    for (int c = 0; c < HID; c++) { atomicAdd(&ssum[c], h[c]); atomicAdd(&ssq[c], h[c] * h[c]); }
    __syncthreads();
    if (tid < HID) {
        float mean = ssum[tid] / (float)GG;
        float var  = ssq[tid] / (float)GG - mean * mean;
        float a = g[tid] * rsqrtf(var + EPS);
        coef[tid] = a; coef[HID + tid] = be[tid] - mean * a;
    }
    __syncthreads();
#pragma unroll
    for (int c = 0; c < HID; c++) h[c] = lrelu(coef[c] * h[c] + coef[HID + c]);
    __syncthreads();
}

__global__ __launch_bounds__(GG) void final_ffn(const float* __restrict__ pooled,
                                                const float* w1, const float* b1,
                                                const float* g1, const float* be1,
                                                const float* w2, const float* b2,
                                                const float* g2, const float* be2,
                                                const float* w3, const float* b3,
                                                float* __restrict__ out) {
    __shared__ float ssum[HID], ssq[HID], coef[2 * HID];
    int tid = threadIdx.x;
    float in[5];
#pragma unroll
    for (int c = 0; c < 5; c++) in[c] = pooled[tid * 16 + c];
    float h[HID];
#pragma unroll
    for (int c = 0; c < HID; c++) {
        float m = b1[c];
#pragma unroll
        for (int k = 0; k < 5; k++) m += in[k] * w1[k * HID + c];
        h[c] = m;
    }
    bn_act_block(h, g1, be1, ssum, ssq, coef);
    float h2[HID];
#pragma unroll
    for (int c = 0; c < HID; c++) {
        float m = b2[c];
#pragma unroll
        for (int k = 0; k < HID; k++) m += h[k] * w2[k * HID + c];
        h2[c] = m;
    }
    bn_act_block(h2, g2, be2, ssum, ssq, coef);
    float o = b3[0];
#pragma unroll
    for (int k = 0; k < HID; k++) o += h2[k] * w3[k];
    out[tid] = o;
}

// ---------------- host orchestration ----------------
extern "C" void kernel_launch(void* const* d_in, const int* in_sizes, int n_in,
                              void* d_out, int out_size, void* d_ws, size_t ws_size,
                              hipStream_t stream) {
    (void)in_sizes; (void)n_in; (void)out_size; (void)ws_size;
    // Input order (setup_inputs insertion order): x, pos,
    //   p1{w1,b1,g1,be1,w2,b2,g2,be2,w3,b3}, p2{...}, p3{...}, p_out{...}, batch
    const float* x   = (const float*)d_in[0];
    const float* pos = (const float*)d_in[1];
    const float* P1[10]; const float* P2[10]; const float* P3[10]; const float* PO[10];
    for (int i = 0; i < 10; i++) {
        P1[i] = (const float*)d_in[2 + i];
        P2[i] = (const float*)d_in[12 + i];
        P3[i] = (const float*)d_in[22 + i];
        PO[i] = (const float*)d_in[32 + i];
    }
    // param index within each dict: 0:w1 1:b1 2:g1 3:be1 4:w2 5:b2 6:g2 7:be2 8:w3 9:b3

    char* wsp = (char*)d_ws;
    auto alloc = [&](size_t bytes) -> void* {
        void* p = (void*)wsp;
        wsp += (bytes + 255) & ~(size_t)255;
        return p;
    };
    f16*   h0p    = (f16*)  alloc((size_t)NN * 32 * 2);
    f16*   h1p    = (f16*)  alloc((size_t)NN * 32 * 2);
    f16*   h2p    = (f16*)  alloc((size_t)NN * 32 * 2);
    float* h2f    = (float*)alloc((size_t)NN * 16 * 4);
    float* h3f    = (float*)alloc((size_t)NN * 16 * 4);
    f16*   u16    = (f16*)  alloc((size_t)NN * 32 * 2);
    f16*   v16    = (f16*)  alloc((size_t)NN * 32 * 2);
    int*   src1   = (int*)  alloc((size_t)EE * 4);
    int*   src2   = (int*)  alloc((size_t)EE * 4);
    float* stats  = (float*)alloc(64 * 4);
    float* coef1  = (float*)alloc(64 * 4);
    float* coef2  = (float*)alloc(64 * 4);
    float* pooled = (float*)alloc((size_t)GG * 16 * 4);
    f16*   wbuf   = (f16*)  alloc(3 * 3584 * 2);

    const float Ef = (float)EE;
    const int gemmBlocks = (NN / 16) / 8;   // 2048 (8 waves/block, 16-node tile/wave)
    const int edgeBlocks = NN / 8;          // 32768

    prep_weights<<<1, 256, 0, stream>>>(P1[0], P1[4], P1[8],
                                        P2[0], P2[4], P2[8],
                                        P3[0], P3[4], P3[8], wbuf);
    build_h0<<<(NN * 32) / 256, 256, 0, stream>>>(x, pos, h0p);
    knn_kernel<<<GG, NPG, 0, stream>>>(pos, 3, 3, src1);

    f16* W = wbuf;
    // ---- EdgeConv 1: h0(4) -> 10, graph = src1 ----
    node_gemm<<<gemmBlocks, 256, 0, stream>>>(h0p, W, u16, v16);
    zero_buf<<<1, 64, 0, stream>>>(stats, 64);
    edge_stats1<<<NN / 256, 256, 0, stream>>>(u16, v16, src1, P1[1], stats);
    bn_finalize<<<1, 32, 0, stream>>>(stats, P1[2], P1[3], Ef, coef1);
    zero_buf<<<1, 64, 0, stream>>>(stats, 64);
    edge_l2_stats<<<edgeBlocks, 160, 0, stream>>>(u16, v16, src1, P1[1], coef1, W, P1[5], stats);
    bn_finalize<<<1, 32, 0, stream>>>(stats, P1[6], P1[7], Ef, coef2);
    edge_l3_aggr<<<edgeBlocks, 160, 0, stream>>>(u16, v16, src1, P1[1], coef1, W, P1[5], coef2,
                                                 P1[9], 10, h1p, nullptr);

    W = wbuf + 3584;
    // ---- EdgeConv 2: h1(10) -> 10, graph = src1 ----
    node_gemm<<<gemmBlocks, 256, 0, stream>>>(h1p, W, u16, v16);
    zero_buf<<<1, 64, 0, stream>>>(stats, 64);
    edge_stats1<<<NN / 256, 256, 0, stream>>>(u16, v16, src1, P2[1], stats);
    bn_finalize<<<1, 32, 0, stream>>>(stats, P2[2], P2[3], Ef, coef1);
    zero_buf<<<1, 64, 0, stream>>>(stats, 64);
    edge_l2_stats<<<edgeBlocks, 160, 0, stream>>>(u16, v16, src1, P2[1], coef1, W, P2[5], stats);
    bn_finalize<<<1, 32, 0, stream>>>(stats, P2[6], P2[7], Ef, coef2);
    edge_l3_aggr<<<edgeBlocks, 160, 0, stream>>>(u16, v16, src1, P2[1], coef1, W, P2[5], coef2,
                                                 P2[9], 10, h2p, h2f);

    // rebuild knn on features
    knn_kernel<<<GG, NPG, 0, stream>>>(h2f, 10, 16, src2);

    W = wbuf + 7168;
    // ---- EdgeConv 3: h2(10) -> 5, graph = src2 ----
    node_gemm<<<gemmBlocks, 256, 0, stream>>>(h2p, W, u16, v16);
    zero_buf<<<1, 64, 0, stream>>>(stats, 64);
    edge_stats1<<<NN / 256, 256, 0, stream>>>(u16, v16, src2, P3[1], stats);
    bn_finalize<<<1, 32, 0, stream>>>(stats, P3[2], P3[3], Ef, coef1);
    zero_buf<<<1, 64, 0, stream>>>(stats, 64);
    edge_l2_stats<<<edgeBlocks, 160, 0, stream>>>(u16, v16, src2, P3[1], coef1, W, P3[5], stats);
    bn_finalize<<<1, 32, 0, stream>>>(stats, P3[6], P3[7], Ef, coef2);
    edge_l3_aggr<<<edgeBlocks, 160, 0, stream>>>(u16, v16, src2, P3[1], coef1, W, P3[5], coef2,
                                                 P3[9], 5, nullptr, h3f);

    // ---- pool + final FFN ----
    pool_kernel<<<GG, NPG, 0, stream>>>(h3f, pooled);
    final_ffn<<<1, GG, 0, stream>>>(pooled,
                                    PO[0], PO[1], PO[2], PO[3],
                                    PO[4], PO[5], PO[6], PO[7],
                                    PO[8], PO[9], (float*)d_out);
}
